// ProbSparseSelfAttention_19421842112999
// MI455X (gfx1250) — matrix operand order repaired and checked by reading, not compiled, not run on hardware
//
#include <hip/hip_runtime.h>
#include <hip/hip_bf16.h>
#include <stdint.h>

// ---------------------------------------------------------------------------
// MHA for MI455X (gfx1250, wave32): all matmuls through
// v_wmma_f32_16x16x32_bf16; A-tile staging via global_load_async_to_lds_b128
// (ASYNCcnt); softmax row-sums folded into an extra WMMA against a constant
// ones-matrix; softmax runs in log2 domain with the 1/sqrt(dh)*log2(e) scale
// folded into the Q projection epilogue; global_prefetch_b8 latency hiding.
// ---------------------------------------------------------------------------

typedef __attribute__((ext_vector_type(16))) __bf16 v16bf;
typedef __attribute__((ext_vector_type(8)))  float  v8f;

#define D_MODEL   1024
#define NHEAD     16
#define HEAD_DIM  64
#define SEQ_LEN   2048
#define BATCH     4
#define MTOT      (BATCH * SEQ_LEN)   // 8192 rows

// 1/sqrt(64) * log2(e): scores come out of QK^T already in log2 units
#define Q_SCALE   0.18033688f

union FragB { v16bf v; uint32_t u[8]; uint4 q[2]; };

__device__ __forceinline__ uint16_t f2bf(float f) {
  // round-to-nearest-even f32 -> bf16
  uint32_t x = __builtin_bit_cast(uint32_t, f);
  uint32_t r = x + 0x7FFFu + ((x >> 16) & 1u);
  return (uint16_t)(r >> 16);
}

// async 16B global -> LDS copy (per lane), tracked by ASYNCcnt
__device__ __forceinline__ void async_g2l_b128(uint32_t lds_byte_addr,
                                               const void* gaddr) {
  asm volatile("global_load_async_to_lds_b128 %0, %1, off"
               :: "v"(lds_byte_addr), "v"((uint64_t)(uintptr_t)gaddr)
               : "memory");
}
__device__ __forceinline__ void wait_async0() {
  asm volatile("s_wait_asynccnt 0" ::: "memory");
}

// ---------------------------------------------------------------------------
// fp32 -> bf16 elementwise cast
// ---------------------------------------------------------------------------
__global__ void cast_f32_bf16(const float* __restrict__ in,
                              uint16_t* __restrict__ out, int n) {
  int i = blockIdx.x * blockDim.x + threadIdx.x;
  int stride = gridDim.x * blockDim.x;
  for (; i < n; i += stride) out[i] = f2bf(in[i]);
}

// ---------------------------------------------------------------------------
// Tiled WMMA GEMM:  Y = X(bf16 [M,K]) @ W(bf16 [K,N]) + bias(f32 [N])
// Block: 256 threads = 8 waves, tile BM=64 x BN=128, BK=32.
// Wave (2x4 grid) owns a 32x32 micro-tile = 2x2 WMMA tiles.
// MODE 0: f32 out, row-major [M,N]              (O projection -> d_out)
// MODE 1: bf16 out, head-major [B,H,L,dh]       (K projection)
// MODE 2: bf16 out, head-major [B,H,dh,L]       (V projection, d-major)
// MODE 3: MODE 1 * Q_SCALE                      (Q projection, pre-scaled)
// ---------------------------------------------------------------------------
template <int MODE>
__global__ __launch_bounds__(256)
void gemm_bf16(const uint16_t* __restrict__ X, const uint16_t* __restrict__ W,
               const float* __restrict__ bias, void* __restrict__ out,
               int M, int N, int K) {
  __shared__ uint16_t As[64 * 32];    // [m][k] row-major
  __shared__ uint16_t Bs[128 * 32];   // [n][k] (transposed: K-pairs contiguous)

  const int tid  = threadIdx.x;
  const int lane = tid & 31;
  const int wave = tid >> 5;
  const int wm   = wave >> 2;          // 0..1
  const int wn   = wave & 3;           // 0..3
  const int blkM = blockIdx.x * 64;
  const int blkN = blockIdx.y * 128;

  v8f acc[2][2];
  for (int a = 0; a < 2; ++a)
    for (int b = 0; b < 2; ++b)
      for (int e = 0; e < 8; ++e) acc[a][b][e] = 0.f;

  const int arow = tid >> 2, acol = (tid & 3) * 8;    // A coop load: 64x32
  const int brow = tid >> 3, bcol = (tid & 7) * 16;   // B coop load: 32x128
  const int lm    = lane & 15;
  const int half0 = (lane < 16) ? 0 : 8;    // A/C lane-half bases (ISA layout)
  const int kb0   = (lane < 16) ? 0 : 16;   // unused: B fragments also use half0

  const uint16_t* aSrcBase = X + (size_t)(blkM + arow) * K + acol;
  const uint16_t* bSrcBase = W + (size_t)brow * N + blkN + bcol;
  const uint32_t aLds = (uint32_t)(uintptr_t)(As + arow * 32 + acol);

  for (int k0 = 0; k0 < K; k0 += 32) {
    // --- stage A tile: async DMA global -> LDS (16B per lane) ---
    async_g2l_b128(aLds, aSrcBase + k0);
    // --- stage B tile transposed into [n][k] (register path) ---
    const uint16_t* wsrc = bSrcBase + (size_t)k0 * N;
    uint4 b0 = *(const uint4*)(wsrc);
    uint4 b1 = *(const uint4*)(wsrc + 8);
    // --- prefetch next K-step tiles into cache while this one computes ---
    if (k0 + 32 < K) {
      __builtin_prefetch(aSrcBase + k0 + 32, 0, 1);
      __builtin_prefetch(bSrcBase + (size_t)(k0 + 32) * N, 0, 1);
    }
    uint32_t bw[8] = {b0.x, b0.y, b0.z, b0.w, b1.x, b1.y, b1.z, b1.w};
#pragma unroll
    for (int i = 0; i < 8; ++i) {
      Bs[(bcol + 2 * i    ) * 32 + brow] = (uint16_t)(bw[i] & 0xFFFFu);
      Bs[(bcol + 2 * i + 1) * 32 + brow] = (uint16_t)(bw[i] >> 16);
    }
    wait_async0();        // A tile landed in LDS
    __syncthreads();

    // --- fragments (ISA 16-bit A/B VGPR layouts) ---
    FragB afr[2], bfr[2];
#pragma unroll
    for (int mt = 0; mt < 2; ++mt) {
      const uint16_t* ap = As + (wm * 32 + mt * 16 + lm) * 32;
      afr[mt].q[0] = *(const uint4*)(ap + half0);
      afr[mt].q[1] = *(const uint4*)(ap + half0 + 16);
    }
#pragma unroll
    for (int nt = 0; nt < 2; ++nt) {
      const uint16_t* bp = Bs + (wn * 32 + nt * 16 + lm) * 32 + half0;
      bfr[nt].q[0] = *(const uint4*)(bp);
      bfr[nt].q[1] = *(const uint4*)(bp + 16);
    }
#pragma unroll
    for (int mt = 0; mt < 2; ++mt)
#pragma unroll
      for (int nt = 0; nt < 2; ++nt)
        acc[mt][nt] = __builtin_amdgcn_wmma_f32_16x16x32_bf16(
            false, afr[mt].v, false, bfr[nt].v, (short)0, acc[mt][nt],
            false, false);
    __syncthreads();
  }

  // --- epilogue: bias (+ optional scale) + compile-time layout store ---
  const float osc = (MODE == 3) ? Q_SCALE : 1.0f;
#pragma unroll
  for (int mt = 0; mt < 2; ++mt) {
#pragma unroll
    for (int nt = 0; nt < 2; ++nt) {
      const int n = blkN + wn * 32 + nt * 16 + lm;
      const float bn = bias[n];
#pragma unroll
      for (int j = 0; j < 8; ++j) {
        const int m = blkM + wm * 32 + mt * 16 + half0 + j;
        const float val = (acc[mt][nt][j] + bn) * osc;
        if (MODE == 0) {
          ((float*)out)[(size_t)m * N + n] = val;
        } else {
          const int bb = m >> 11, l = m & 2047, h = n >> 6, d = n & 63;
          size_t idx;
          if (MODE == 1 || MODE == 3)
            idx = ((size_t)(bb * NHEAD + h)) * (SEQ_LEN * HEAD_DIM) +
                  (size_t)l * HEAD_DIM + d;
          else
            idx = ((size_t)(bb * NHEAD + h)) * (SEQ_LEN * HEAD_DIM) +
                  (size_t)d * SEQ_LEN + l;
          ((uint16_t*)out)[idx] = f2bf(val);
        }
      }
    }
  }
}

// ---------------------------------------------------------------------------
// Flash-style attention.  One wave handles 16 query rows of one (b,h).
// Qp (pre-scaled by Q_SCALE) / Kp: bf16 [B*H, L, dh]; Vt: bf16 [B*H, dh, L];
// ctx: bf16 [B,L,D].  Scores come out of WMMA already in log2 units, so the
// online softmax uses native-base exp2f.  Row-max via 16-lane shfl butterfly;
// row-sum via WMMA P @ ones; P transposed C->A layout through per-wave LDS.
// ---------------------------------------------------------------------------
__global__ __launch_bounds__(128)
void attn_kernel(const uint16_t* __restrict__ Qp, const uint16_t* __restrict__ Kp,
                 const uint16_t* __restrict__ Vt, uint16_t* __restrict__ ctx) {
  __shared__ uint16_t Plds[4][16 * 32];   // per-wave P staging

  const int lane = threadIdx.x & 31;
  const int wave = threadIdx.x >> 5;
  const int q0 = blockIdx.x * 64 + wave * 16;
  const int bh = blockIdx.y;                 // b*NHEAD + h
  const uint16_t* Qb = Qp + (size_t)bh * SEQ_LEN * HEAD_DIM;
  const uint16_t* Kb = Kp + (size_t)bh * SEQ_LEN * HEAD_DIM;
  const uint16_t* Vb = Vt + (size_t)bh * HEAD_DIM * SEQ_LEN;

  const int lm    = lane & 15;
  const int half0 = (lane < 16) ? 0 : 8;
  const int kb0   = (lane < 16) ? 0 : 16;

  // Q A-fragments, d = [0,32) and [32,64): 2 x b128 loads each
  FragB aq[2];
#pragma unroll
  for (int t = 0; t < 2; ++t) {
    const uint16_t* qp = Qb + (size_t)(q0 + lm) * HEAD_DIM + t * 32 + half0;
    aq[t].q[0] = *(const uint4*)(qp);
    aq[t].q[1] = *(const uint4*)(qp + 16);
  }

  // constant ones B-fragment (bf16 1.0 pairs) for WMMA row-sum
  FragB onesF;
#pragma unroll
  for (int i = 0; i < 8; ++i) onesF.u[i] = 0x3F803F80u;

  float rowmax[8], rowsum[8];
  v8f accv[4];
  for (int j = 0; j < 8; ++j) { rowmax[j] = -1e30f; rowsum[j] = 0.f; }
  for (int t = 0; t < 4; ++t)
    for (int e = 0; e < 8; ++e) accv[t][e] = 0.f;

  uint16_t* Pw = &Plds[wave][0];

  for (int kb = 0; kb < SEQ_LEN; kb += 32) {
    // ---- scores (log2 units): two 16x16 tiles over 32 keys ----
    v8f s[2];
#pragma unroll
    for (int nt = 0; nt < 2; ++nt) {
      for (int e = 0; e < 8; ++e) s[nt][e] = 0.f;
      const uint16_t* kp = Kb + (size_t)(kb + nt * 16 + lm) * HEAD_DIM + half0;
      FragB bk;
#pragma unroll
      for (int t = 0; t < 2; ++t) {
        bk.q[0] = *(const uint4*)(kp + t * 32);
        bk.q[1] = *(const uint4*)(kp + t * 32 + 16);
        s[nt] = __builtin_amdgcn_wmma_f32_16x16x32_bf16(
            false, aq[t].v, false, bk.v, (short)0, s[nt], false, false);
      }
    }

    // ---- online softmax in log2 domain: row-max butterfly + rescale ----
    float fac[8], p0[8], p1[8];
#pragma unroll
    for (int j = 0; j < 8; ++j) {
      float v0 = s[0][j];
      float v1 = s[1][j];
      float mx = fmaxf(v0, v1);
#pragma unroll
      for (int off = 1; off < 16; off <<= 1)
        mx = fmaxf(mx, __shfl_xor(mx, off, 32));
      const float mnew = fmaxf(rowmax[j], mx);
      fac[j] = exp2f(rowmax[j] - mnew);     // native v_exp_f32
      rowmax[j] = mnew;
      p0[j] = exp2f(v0 - mnew);
      p1[j] = exp2f(v1 - mnew);
#pragma unroll
      for (int t = 0; t < 4; ++t) accv[t][j] *= fac[j];
    }

    // ---- P: C-layout -> A-layout via per-wave LDS (in-order within wave) ----
#pragma unroll
    for (int j = 0; j < 8; ++j) {
      const int m = half0 + j;
      Pw[m * 32 + lm]      = f2bf(p0[j]);
      Pw[m * 32 + 16 + lm] = f2bf(p1[j]);
    }
    FragB aP;
    {
      const uint16_t* pp = Pw + lm * 32 + half0;
      aP.q[0] = *(const uint4*)(pp);
      aP.q[1] = *(const uint4*)(pp + 16);
    }

    // ---- row-sum of P via WMMA against ones (every C column = row sum) ----
    v8f zf;
    for (int e = 0; e < 8; ++e) zf[e] = 0.f;
    v8f psum = __builtin_amdgcn_wmma_f32_16x16x32_bf16(
        false, aP.v, false, onesF.v, (short)0, zf, false, false);
#pragma unroll
    for (int j = 0; j < 8; ++j)
      rowsum[j] = rowsum[j] * fac[j] + psum[j];

    // ---- O += P @ V : V is d-major, so K-pairs are contiguous b128 loads ----
#pragma unroll
    for (int nt = 0; nt < 4; ++nt) {
      const uint16_t* vp = Vb + (size_t)(nt * 16 + lm) * SEQ_LEN + kb + half0;
      FragB bv;
      bv.q[0] = *(const uint4*)(vp);
      bv.q[1] = *(const uint4*)(vp + 16);
      accv[nt] = __builtin_amdgcn_wmma_f32_16x16x32_bf16(
          false, aP.v, false, bv.v, (short)0, accv[nt], false, false);
    }
  }

  // ---- normalize + store ctx (bf16 [B, L, D], head-concat) ----
  const int b = bh >> 4, h = bh & 15;
#pragma unroll
  for (int j = 0; j < 8; ++j) {
    const float inv = 1.0f / rowsum[j];
    const int m = q0 + half0 + j;
#pragma unroll
    for (int nt = 0; nt < 4; ++nt) {
      const int d = h * HEAD_DIM + nt * 16 + lm;
      ctx[((size_t)(b * SEQ_LEN + m)) * D_MODEL + d] = f2bf(accv[nt][j] * inv);
    }
  }
}

// ---------------------------------------------------------------------------
extern "C" void kernel_launch(void* const* d_in, const int* in_sizes, int n_in,
                              void* d_out, int out_size, void* d_ws, size_t ws_size,
                              hipStream_t stream) {
  (void)in_sizes; (void)n_in; (void)out_size; (void)ws_size;
  const float* q  = (const float*)d_in[0];
  const float* k  = (const float*)d_in[1];
  const float* v  = (const float*)d_in[2];
  const float* Wq = (const float*)d_in[3];
  const float* bq = (const float*)d_in[4];
  const float* Wk = (const float*)d_in[5];
  const float* bk = (const float*)d_in[6];
  const float* Wv = (const float*)d_in[7];
  const float* bv = (const float*)d_in[8];
  const float* Wo = (const float*)d_in[9];
  const float* bo = (const float*)d_in[10];

  uint8_t* ws = (uint8_t*)d_ws;
  const size_t MB = 1024u * 1024u;
  uint16_t* Qb  = (uint16_t*)(ws +   0 * MB);   // bf16 inputs  [8192,1024]
  uint16_t* Kb  = (uint16_t*)(ws +  16 * MB);
  uint16_t* Vb  = (uint16_t*)(ws +  32 * MB);
  uint16_t* Wqb = (uint16_t*)(ws +  48 * MB);   // bf16 weights [1024,1024]
  uint16_t* Wkb = (uint16_t*)(ws +  50 * MB);
  uint16_t* Wvb = (uint16_t*)(ws +  52 * MB);
  uint16_t* Wob = (uint16_t*)(ws +  54 * MB);
  uint16_t* Qp  = (uint16_t*)(ws +  56 * MB);   // [B,H,L,dh] (pre-scaled)
  uint16_t* Kp  = (uint16_t*)(ws +  72 * MB);   // [B,H,L,dh]
  uint16_t* Vpt = (uint16_t*)(ws +  88 * MB);   // [B,H,dh,L]
  uint16_t* ctx = (uint16_t*)(ws + 104 * MB);   // [B,L,D]

  const int nQKV = MTOT * D_MODEL;        // 8,388,608
  const int nW   = D_MODEL * D_MODEL;     // 1,048,576
  cast_f32_bf16<<<4096, 256, 0, stream>>>(q,  Qb,  nQKV);
  cast_f32_bf16<<<4096, 256, 0, stream>>>(k,  Kb,  nQKV);
  cast_f32_bf16<<<4096, 256, 0, stream>>>(v,  Vb,  nQKV);
  cast_f32_bf16<<<1024, 256, 0, stream>>>(Wq, Wqb, nW);
  cast_f32_bf16<<<1024, 256, 0, stream>>>(Wk, Wkb, nW);
  cast_f32_bf16<<<1024, 256, 0, stream>>>(Wv, Wvb, nW);
  cast_f32_bf16<<<1024, 256, 0, stream>>>(Wo, Wob, nW);

  dim3 gblk(MTOT / 64, D_MODEL / 128);    // (128, 8)
  gemm_bf16<3><<<gblk, 256, 0, stream>>>(Qb, Wqb, bq, Qp,  MTOT, D_MODEL, D_MODEL);
  gemm_bf16<1><<<gblk, 256, 0, stream>>>(Kb, Wkb, bk, Kp,  MTOT, D_MODEL, D_MODEL);
  gemm_bf16<2><<<gblk, 256, 0, stream>>>(Vb, Wvb, bv, Vpt, MTOT, D_MODEL, D_MODEL);

  attn_kernel<<<dim3(SEQ_LEN / 64, BATCH * NHEAD), 128, 0, stream>>>(Qp, Kp, Vpt, ctx);

  gemm_bf16<0><<<gblk, 256, 0, stream>>>(ctx, Wob, bo, d_out, MTOT, D_MODEL, D_MODEL);
}
